// LinearCorssAttention_10282151707301
// MI455X (gfx1250) — matrix operand order repaired and checked by reading, not compiled, not run on hardware
//
#include <hip/hip_runtime.h>

// ---------------------------------------------------------------------------
// LinearCrossAttention, MI455X (gfx1250, wave32, WMMA)
//
// Algebra: out = W_eff[b] @ x[b] + bo, with
//   W_eff = SCALE * sum_h (Wo_h @ ctx_h^T) @ Wq_h
//   ctx_h[d,e] = (1/Z[d]) * sum_c Wv[e',c] * G[d,c],  e' = 32h+e
//   G[d,c] = sum_n exp(K[d,n]) C[c,n],  Z[d] = G[d,128] (ones-column trick)
//   K = Wk @ C   (softmax max-subtraction skipped: logits are O(1))
// Memory traffic: context once + x once + out once ~= 402 MB (roofline min).
// ---------------------------------------------------------------------------

typedef __attribute__((ext_vector_type(16))) __bf16 v16bf;
typedef __attribute__((ext_vector_type(8)))  float  v8f;

#define HEADS     4
#define DHEAD     32
#define DIM       128
#define NTOT      65536        // 256*256
#define NCHUNK    512          // spatial columns per block
#define TCOL      64           // columns per inner tile
#define NITER     (NCHUNK / TCOL)
#define SCALE_F   0.17677669529663689f   // 32^-0.5
#define GROWS     144          // 128 channels + ones-col(128) + pad

union Frag16 { unsigned u[8]; v16bf v; };

// float -> bf16 (RNE), bit-exact, no reliance on scalar __bf16 arithmetic
__device__ __forceinline__ unsigned short f2bf(float f) {
  unsigned u = __float_as_uint(f);
  unsigned r = u + 0x7FFFu + ((u >> 16) & 1u);
  return (unsigned short)(r >> 16);
}

// A-matrix fragment, 16x32 bf16 tile at (row0, k0) from LDS row-major [m][k],
// pitch in elements (even). Per ISA 7.12.2: lane L (L<16) row M=L holds
// K = {0..7, 16..23}; lane L+16 holds K = {8..15, 24..31}; 2 bf16 per VGPR.
__device__ __forceinline__ v16bf load_frag_A(const unsigned short* base,
                                             int pitch, int row0, int k0,
                                             int lane) {
  const unsigned* p32 = reinterpret_cast<const unsigned*>(base);
  const int half = (lane >> 4) & 1;
  const int rowoff = (row0 + (lane & 15)) * pitch + k0;
  Frag16 f;
#pragma unroll
  for (int j = 0; j < 8; ++j) {
    int kk = ((j & 4) << 2) + (half << 3) + ((j & 3) << 1);
    f.u[j] = p32[(rowoff + kk) >> 1];
  }
  return f.v;
}

// B-matrix fragment, 32x16 bf16 tile at (k0, n0). Source LDS is laid out
// [n][k] (k contiguous), pitch in elements (even). Same K order as A: lanes
// 0-15 carry K = {0..7, 16..23}; lanes 16-31 carry K = {8..15, 24..31}.
__device__ __forceinline__ v16bf load_frag_B(const unsigned short* base,
                                             int pitch, int n0, int k0,
                                             int lane) {
  const unsigned* p32 = reinterpret_cast<const unsigned*>(base);
  const int half = (lane >> 4) & 1;
  const int rowoff = (n0 + (lane & 15)) * pitch + k0;
  Frag16 f;
#pragma unroll
  for (int j = 0; j < 8; ++j) {
    int kk = ((j & 4) << 2) + (half << 3) + ((j & 3) << 1);
    f.u[j] = p32[(rowoff + kk) >> 1];
  }
  return f.v;
}

__global__ void zero_kernel(float* __restrict__ p, int n) {
  int i = blockIdx.x * 256 + threadIdx.x;
  if (i < n) p[i] = 0.0f;
}

// ---------------------------------------------------------------------------
// Kernel A: one streaming pass over `context`.
//   E = exp(Wk @ C_tile)  (WMMA bf16, fp32 accum, v_exp_f32)
//   G += E @ C_tile^T     (WMMA bf16; 9th c-tile = ones column -> Z)
// ---------------------------------------------------------------------------
__global__ __launch_bounds__(256) void ctx_accum_kernel(
    const float* __restrict__ ctxt, const float* __restrict__ Wk,
    float* __restrict__ gws) {
  __shared__ __align__(16) unsigned short s_wk[DIM * DIM];      // Wk  [row][c]
  __shared__ __align__(16) unsigned short s_ct[GROWS * TCOL];   // C   [c][col]
  __shared__ __align__(16) unsigned short s_ctT[TCOL * DIM];    // C^T [col][c]
  __shared__ __align__(16) unsigned short s_e[DIM * TCOL];      // E   [d][col]

  const int tid  = threadIdx.x;
  const int lane = tid & 31;
  const int w    = tid >> 5;           // wave id 0..7
  const int half = (lane >> 4) & 1;
  const int n15  = lane & 15;
  const int bat  = blockIdx.y;
  const int chunk = blockIdx.x;

  // stage Wk (fp32 -> bf16), float4 loads
  for (int i = tid; i < DIM * DIM / 4; i += 256) {
    const float4 v = reinterpret_cast<const float4*>(Wk)[i];
    s_wk[i * 4 + 0] = f2bf(v.x);
    s_wk[i * 4 + 1] = f2bf(v.y);
    s_wk[i * 4 + 2] = f2bf(v.z);
    s_wk[i * 4 + 3] = f2bf(v.w);
  }
  // constant tail rows of s_ct: row 128 = ones (gives Z), rows 129..143 = 0
  for (int i = tid; i < 16 * TCOL; i += 256) {
    int row = i / TCOL;
    s_ct[(DIM + row) * TCOL + (i % TCOL)] =
        (row == 0) ? f2bf(1.0f) : (unsigned short)0;
  }
  __syncthreads();

  // Wk A-fragments for this wave's 16-row band (constant across tiles)
  v16bf awk[4];
#pragma unroll
  for (int ks = 0; ks < 4; ++ks)
    awk[ks] = load_frag_A(s_wk, DIM, 16 * w, 32 * ks, lane);

  // G accumulators: wave w owns d-rows [16w,16w+16), 9 c-tiles (144 cols)
  v8f zf = {};
  v8f g[9];
#pragma unroll
  for (int t = 0; t < 9; ++t) g[t] = zf;

  const float* cbase = ctxt + (bat * DIM) * NTOT + chunk * NCHUNK;

  for (int it = 0; it < NITER; ++it) {
    const float* tbase = cbase + it * TCOL;
    // stage 128x64 context tile in both layouts (fp32 -> bf16)
    for (int i = tid; i < DIM * (TCOL / 4); i += 256) {
      int row = i >> 4;                  // channel 0..127
      int qc  = i & 15;                  // float4 column group
      const float4 v =
          *reinterpret_cast<const float4*>(tbase + row * NTOT + qc * 4);
      const int col = qc * 4;
      unsigned short b0 = f2bf(v.x), b1 = f2bf(v.y), b2 = f2bf(v.z),
                     b3 = f2bf(v.w);
      s_ct[row * TCOL + col + 0] = b0;
      s_ct[row * TCOL + col + 1] = b1;
      s_ct[row * TCOL + col + 2] = b2;
      s_ct[row * TCOL + col + 3] = b3;
      s_ctT[(col + 0) * DIM + row] = b0;
      s_ctT[(col + 1) * DIM + row] = b1;
      s_ctT[(col + 2) * DIM + row] = b2;
      s_ctT[(col + 3) * DIM + row] = b3;
    }
    __syncthreads();

    // GEMM1: E rows [16w,16w+16) for all 64 cols; wave-local E band, so no
    // barrier needed before GEMM2 (this wave reads only the rows it wrote).
#pragma unroll
    for (int nt = 0; nt < 4; ++nt) {
      v8f acc = {};
#pragma unroll
      for (int ks = 0; ks < 4; ++ks) {
        v16bf bf = load_frag_B(s_ctT, DIM, 16 * nt, 32 * ks, lane);
        acc = __builtin_amdgcn_wmma_f32_16x16x32_bf16(
            false, awk[ks], false, bf, (short)0, acc, false, false);
      }
      const int ncol = nt * 16 + n15;
#pragma unroll
      for (int r = 0; r < 8; ++r) {
        int m = 16 * w + r + half * 8;
        s_e[m * TCOL + ncol] = f2bf(__expf(acc[r]));
      }
    }

    // GEMM2: G[16w..16w+15][0..143] += E_band @ C_tile^T (c-dim = N)
    v16bf ae[2];
#pragma unroll
    for (int ks = 0; ks < 2; ++ks)
      ae[ks] = load_frag_A(s_e, TCOL, 16 * w, 32 * ks, lane);
#pragma unroll
    for (int ct = 0; ct < 9; ++ct) {
#pragma unroll
      for (int ks = 0; ks < 2; ++ks) {
        v16bf bf = load_frag_B(s_ct, TCOL, 16 * ct, 32 * ks, lane);
        g[ct] = __builtin_amdgcn_wmma_f32_16x16x32_bf16(
            false, ae[ks], false, bf, (short)0, g[ct], false, false);
      }
    }
    __syncthreads();   // protect s_ct/s_ctT before next staging
  }

  // flush partial G with native fp32 global atomics (no-return form)
  float* gb = gws + bat * DIM * GROWS;
#pragma unroll
  for (int ct = 0; ct < 9; ++ct) {
#pragma unroll
    for (int r = 0; r < 8; ++r) {
      int d = 16 * w + r + half * 8;
      int c = 16 * ct + n15;
      __hip_atomic_fetch_add(gb + d * GROWS + c, g[ct][r], __ATOMIC_RELAXED,
                             __HIP_MEMORY_SCOPE_AGENT);
    }
  }
}

// ---------------------------------------------------------------------------
// Kernel B: tiny per-batch fold of Wv, 1/Z, Wo, Wq into W_eff (128x128).
// ---------------------------------------------------------------------------
__global__ __launch_bounds__(256) void combine_kernel(
    const float* __restrict__ Wq, const float* __restrict__ Wv,
    const float* __restrict__ Wo, const float* __restrict__ gws,
    float* __restrict__ weff) {
  __shared__ float s_zi[DIM];
  __shared__ float s_sd[DIM * DHEAD];   // ctx_norm[g][e] = S[g,e']/Z[g]
  __shared__ float s_T[DIM * DIM];      // T[o][g]
  const int tid = threadIdx.x;
  const int bat = blockIdx.x;
  const float* G = gws + bat * DIM * GROWS;

  if (tid < DIM) s_zi[tid] = 1.0f / G[tid * GROWS + DIM];
  __syncthreads();

  for (int i = tid; i < DIM * DHEAD; i += 256) {
    int gr = i >> 5, e = i & 31, h = gr >> 5;
    const float* wvrow = Wv + (h * DHEAD + e) * DIM;
    const float* grow  = G + gr * GROWS;
    float s = 0.0f;
    for (int c = 0; c < DIM; ++c) s += wvrow[c] * grow[c];
    s_sd[i] = s * s_zi[gr];
  }
  __syncthreads();

  for (int i = tid; i < DIM * DIM; i += 256) {
    int o = i >> 7, gr = i & 127, h = gr >> 5;
    const float* worow = Wo + o * DIM + h * DHEAD;
    const float* sd = s_sd + gr * DHEAD;
    float s = 0.0f;
    for (int e = 0; e < DHEAD; ++e) s += worow[e] * sd[e];
    s_T[i] = s;
  }
  __syncthreads();

  float* wout = weff + bat * DIM * DIM;
  for (int i = tid; i < DIM * DIM; i += 256) {
    int o = i >> 7, c = i & 127;
    const float* tr = s_T + o * DIM;
    float s = 0.0f;
    for (int gr = 0; gr < DIM; ++gr) s += tr[gr] * Wq[gr * DIM + c];
    wout[i] = s * SCALE_F;
  }
}

// ---------------------------------------------------------------------------
// Kernel C: out[b] = W_eff[b] @ x[b] + bo  (one streaming pass over x)
// ---------------------------------------------------------------------------
__global__ __launch_bounds__(256) void out_gemm_kernel(
    const float* __restrict__ x, const float* __restrict__ weff,
    const float* __restrict__ bo, float* __restrict__ out) {
  __shared__ __align__(16) unsigned short s_w[DIM * DIM];     // W_eff [o][c]
  __shared__ __align__(16) unsigned short s_xT[TCOL * DIM];   // x^T  [col][c]

  const int tid  = threadIdx.x;
  const int lane = tid & 31;
  const int w    = tid >> 5;
  const int half = (lane >> 4) & 1;
  const int n15  = lane & 15;
  const int bat  = blockIdx.y;
  const int chunk = blockIdx.x;

  const float* wb = weff + bat * DIM * DIM;
  for (int i = tid; i < DIM * DIM / 4; i += 256) {
    const float4 v = reinterpret_cast<const float4*>(wb)[i];
    s_w[i * 4 + 0] = f2bf(v.x);
    s_w[i * 4 + 1] = f2bf(v.y);
    s_w[i * 4 + 2] = f2bf(v.z);
    s_w[i * 4 + 3] = f2bf(v.w);
  }
  __syncthreads();

  v16bf aw[4];
#pragma unroll
  for (int ks = 0; ks < 4; ++ks)
    aw[ks] = load_frag_A(s_w, DIM, 16 * w, 32 * ks, lane);
  float bor[8];
#pragma unroll
  for (int r = 0; r < 8; ++r) bor[r] = bo[16 * w + r + half * 8];

  const float* xb = x + (bat * DIM) * NTOT + chunk * NCHUNK;
  float* ob = out + (bat * DIM) * NTOT + chunk * NCHUNK;

  for (int it = 0; it < NITER; ++it) {
    const float* tb = xb + it * TCOL;
    for (int i = tid; i < DIM * (TCOL / 4); i += 256) {
      int row = i >> 4;
      int qc  = i & 15;
      const float4 v =
          *reinterpret_cast<const float4*>(tb + row * NTOT + qc * 4);
      const int col = qc * 4;
      s_xT[(col + 0) * DIM + row] = f2bf(v.x);
      s_xT[(col + 1) * DIM + row] = f2bf(v.y);
      s_xT[(col + 2) * DIM + row] = f2bf(v.z);
      s_xT[(col + 3) * DIM + row] = f2bf(v.w);
    }
    __syncthreads();

#pragma unroll
    for (int nt = 0; nt < 4; ++nt) {
      v8f acc = {};
#pragma unroll
      for (int ks = 0; ks < 4; ++ks) {
        v16bf bf = load_frag_B(s_xT, DIM, 16 * nt, 32 * ks, lane);
        acc = __builtin_amdgcn_wmma_f32_16x16x32_bf16(
            false, aw[ks], false, bf, (short)0, acc, false, false);
      }
      const int ncol = it * TCOL + nt * 16 + n15;
#pragma unroll
      for (int r = 0; r < 8; ++r) {
        int m = 16 * w + r + half * 8;
        ob[m * NTOT + ncol] = acc[r] + bor[r];
      }
    }
    __syncthreads();
  }
}

// ---------------------------------------------------------------------------
extern "C" void kernel_launch(void* const* d_in, const int* in_sizes, int n_in,
                              void* d_out, int out_size, void* d_ws,
                              size_t ws_size, hipStream_t stream) {
  const float* x    = (const float*)d_in[0];
  const float* ctxt = (const float*)d_in[1];
  const float* Wq   = (const float*)d_in[2];
  const float* Wk   = (const float*)d_in[3];
  const float* Wv   = (const float*)d_in[4];
  const float* Wo   = (const float*)d_in[5];
  const float* bo   = (const float*)d_in[6];
  float* out = (float*)d_out;

  // workspace: G[4][128][144] fp32, then W_eff[4][128][128] fp32 (~544 KB)
  float* gws  = (float*)d_ws;
  float* weff = gws + 4 * DIM * GROWS;

  const int gtot = 4 * DIM * GROWS;  // 73728
  zero_kernel<<<(gtot + 255) / 256, 256, 0, stream>>>(gws, gtot);
  ctx_accum_kernel<<<dim3(NTOT / NCHUNK, 4), 256, 0, stream>>>(ctxt, Wk, gws);
  combine_kernel<<<4, 256, 0, stream>>>(Wq, Wv, Wo, gws, weff);
  out_gemm_kernel<<<dim3(NTOT / NCHUNK, 4), 256, 0, stream>>>(x, weff, bo, out);
}
